// NodeSchNet_3934190044228
// MI455X (gfx1250) — compile-verified, matrix operand order checked by reading, not run on hardware
//
#include <hip/hip_runtime.h>
#include <hip/hip_bf16.h>

typedef __attribute__((ext_vector_type(16))) _Float16 v16h;
typedef __attribute__((ext_vector_type(8)))  float    v8f;
typedef _Float16 h16;

#define NN   10000
#define EE   320000
#define HH   128
#define GG   50
#define GPAD 64
#define LL   6

// workspace layout (bytes)
#define O_GAUSS ((size_t)0)
#define O_C     (O_GAUSS + (size_t)EE*GPAD*2)   // f16 [E,64]
#define O_X     (O_C     + (size_t)EE*4)        // f32 [E]
#define O_AGG   (O_X     + (size_t)NN*HH*4)     // f32 [N,128]
#define O_T     (O_AGG   + (size_t)NN*HH*4)     // f32 [N,128]
#define O_W     (O_T     + (size_t)NN*HH*2)     // f16 [N,128]
// per-layer converted-weight block (element == half), offsets in halfs
#define WLAYER  73728
#define W_W1T   0       // [128][64]   (f,g)   from mlp_w1[g][f], g>=50 -> 0
#define W_W2T   8192    // [128][128]  (fo,fi) from mlp_w2[fi][fo]
#define W_L1T   24576   // [128][128]  (f,h)   from lin1_w[h][f]
#define W_L2T   40960   // [128][128]  (ho,fi) from lin2_w[fi][ho]
#define W_LT    57344   // [128][128]  (ho,hi) from lin_w[hi][ho]

__device__ __forceinline__ int lane_id() { return (int)(threadIdx.x & 31u); }

// 16x32 f16 A/B fragment from row-major [16+][stride] half array.
// lane: row = lane&15; K halves 0..7 -> K k0+{0..7 | +8}, halves 8..15 -> K k0+16+{...}
__device__ __forceinline__ v16h frag_h(const h16* base, int stride, int k0) {
  int l = lane_id(); int r = l & 15; int kh = (l >> 4) << 3;
  const h16* p = base + r * stride + k0 + kh;
  v16h f;
#pragma unroll
  for (int v = 0; v < 4; ++v) {
    f[2*v]   = p[2*v];
    f[2*v+1] = p[2*v+1];
    f[8+2*v] = p[16+2*v];
    f[9+2*v] = p[17+2*v];
  }
  return f;
}

// same fragment, converting from f32 source on the fly
__device__ __forceinline__ v16h frag_f32(const float* base, int stride, int k0) {
  int l = lane_id(); int r = l & 15; int kh = (l >> 4) << 3;
  const float* p = base + r * stride + k0 + kh;
  v16h f;
#pragma unroll
  for (int v = 0; v < 4; ++v) {
    f[2*v]   = (h16)p[2*v];
    f[2*v+1] = (h16)p[2*v+1];
    f[8+2*v] = (h16)p[16+2*v];
    f[9+2*v] = (h16)p[17+2*v];
  }
  return f;
}

__device__ __forceinline__ v8f wmma32(v16h a, v16h b, v8f c) {
  return __builtin_amdgcn_wmma_f32_16x16x32_f16(false, a, false, b, (short)0, c,
                                                false, false);
}

// shifted softplus: log(1+e^x) - log 2, overflow-safe
__device__ __forceinline__ float sspf(float x) {
  float t = __expf(-fabsf(x));
  return fmaxf(x, 0.0f) + __logf(1.0f + t) - 0.69314718056f;
}

// ---------------- one-time kernels ----------------

__global__ void k_prep(const float* __restrict__ pos, const int* __restrict__ ei,
                       h16* __restrict__ gauss, float* __restrict__ Cb) {
  int e = blockIdx.x * blockDim.x + threadIdx.x;
  if (e >= EE) return;
  int r = ei[e], c = ei[EE + e];
  float dx = pos[r*3+0] - pos[c*3+0];
  float dy = pos[r*3+1] - pos[c*3+1];
  float dz = pos[r*3+2] - pos[c*3+2];
  float d  = sqrtf(dx*dx + dy*dy + dz*dz);
  Cb[e] = 0.5f * (__cosf(d * 0.31415926535897932f) + 1.0f);
  const float step  = 10.0f / 49.0f;
  const float coeff = -0.5f / (step * step);
  h16* g = gauss + (size_t)e * GPAD;
  for (int i = 0; i < GPAD; ++i) {
    float dd = d - step * (float)i;
    float v  = (i < GG) ? __expf(coeff * dd * dd) : 0.0f;
    g[i] = (h16)v;
  }
}

__global__ void k_embed(const int* __restrict__ z, const float* __restrict__ emb,
                        float* __restrict__ h) {
  int i = blockIdx.x * blockDim.x + threadIdx.x;
  if (i >= NN * HH) return;
  int n = i >> 7, c = i & 127;
  h[i] = emb[z[n] * HH + c];
}

__global__ void k_convw(const float* __restrict__ w1, const float* __restrict__ w2,
                        const float* __restrict__ l1, const float* __restrict__ l2,
                        const float* __restrict__ lw, h16* __restrict__ wf) {
  int i = blockIdx.x * blockDim.x + threadIdx.x;
  if (i >= LL * WLAYER) return;
  int layer = i / WLAYER;
  int j = i - layer * WLAYER;
  float v;
  if (j < W_W2T)      { int f = j >> 6, g = j & 63;
                        v = (g < GG) ? w1[layer*GG*HH + g*HH + f] : 0.0f; }
  else if (j < W_L1T) { int k = j - W_W2T; int fo = k >> 7, fi = k & 127;
                        v = w2[layer*HH*HH + fi*HH + fo]; }
  else if (j < W_L2T) { int k = j - W_L1T; int f = k >> 7, hh = k & 127;
                        v = l1[layer*HH*HH + hh*HH + f]; }
  else if (j < W_LT)  { int k = j - W_L2T; int ho = k >> 7, fi = k & 127;
                        v = l2[layer*HH*HH + fi*HH + ho]; }
  else                { int k = j - W_LT;  int ho = k >> 7, hi = k & 127;
                        v = lw[layer*HH*HH + hi*HH + ho]; }
  wf[i] = (h16)v;
}

// ---------------- per-layer kernels ----------------

// x = h @ lin1_w (no bias); also zero agg. 256 thr = 8 waves, 1 node-tile each.
__global__ void k_lin1(const float* __restrict__ h, const h16* __restrict__ wf16,
                       float* __restrict__ x, float* __restrict__ agg, int layer) {
  __shared__ h16 sW[HH * HH];
  { const uint32_t* s = (const uint32_t*)(wf16 + (size_t)layer*WLAYER + W_L1T);
    uint32_t* d = (uint32_t*)sW;
    for (int i = threadIdx.x; i < HH*HH/2; i += blockDim.x) d[i] = s[i]; }
  __syncthreads();
  int wid = threadIdx.x >> 5;
  int t = blockIdx.x * 8 + wid;
  if (t >= NN / 16) return;
  for (int i = lane_id(); i < 16 * HH; i += 32) agg[(size_t)t*16*HH + i] = 0.0f;
  const float* A = h + (size_t)t * 16 * HH;
  v16h af[4];
#pragma unroll
  for (int k = 0; k < 4; ++k) af[k] = frag_f32(A, HH, k * 32);
  int col = lane_id() & 15;
  int kh3 = (lane_id() >> 4) << 3;
#pragma unroll
  for (int n0 = 0; n0 < HH; n0 += 16) {
    v8f acc = {};
#pragma unroll
    for (int k = 0; k < 4; ++k)
      acc = wmma32(af[k], frag_h(sW + n0 * HH, HH, k * 32), acc);
#pragma unroll
    for (int r = 0; r < 8; ++r)
      x[((size_t)t*16 + r + kh3) * HH + n0 + col] = acc[r];
  }
}

// fused edge filter + gather + modulate + scatter-add. 128 thr = 4 waves.
__global__ void k_edge(const h16* __restrict__ gauss, const float* __restrict__ Cb,
                       const int* __restrict__ ei, const float* __restrict__ x,
                       const h16* __restrict__ wf16,
                       const float* __restrict__ b1, const float* __restrict__ b2,
                       float* __restrict__ agg, int layer) {
  __shared__ h16 sW2[HH * HH];
  __shared__ h16 sH[4][16 * HH];
  __shared__ int   sRow[4][16];
  __shared__ int   sCol[4][16];
  __shared__ float sCv[4][16];
  const h16* wl = wf16 + (size_t)layer * WLAYER;
  { const uint32_t* s = (const uint32_t*)(wl + W_W2T);
    uint32_t* d = (uint32_t*)sW2;
    for (int i = threadIdx.x; i < HH*HH/2; i += blockDim.x) d[i] = s[i]; }
  int wid = threadIdx.x >> 5;
  int t = blockIdx.x * 4 + wid;            // edge tile (exactly E/16 tiles)
  int l = lane_id();
  if (l < 16) {
    int e = t * 16 + l;
    sRow[wid][l] = ei[e];
    sCol[wid][l] = ei[EE + e];
    sCv[wid][l]  = Cb[e];
  }
  __syncthreads();
  int col = l & 15;
  int kh3 = (l >> 4) << 3;
  const float* b1l = b1 + layer * HH;
  const float* b2l = b2 + layer * HH;
  // GEMM1: hidden = ssp(gauss[16,64] @ w1[64,128] + b1) -> LDS f16
  const h16* A = gauss + (size_t)t * 16 * GPAD;
  v16h ga0 = frag_h(A, GPAD, 0);
  v16h ga1 = frag_h(A, GPAD, 32);
  const h16* W1 = wl + W_W1T;
#pragma unroll
  for (int n0 = 0; n0 < HH; n0 += 16) {
    v8f acc = {};
    acc = wmma32(ga0, frag_h(W1 + n0 * GPAD, GPAD, 0),  acc);
    acc = wmma32(ga1, frag_h(W1 + n0 * GPAD, GPAD, 32), acc);
    float bias = b1l[n0 + col];
#pragma unroll
    for (int r = 0; r < 8; ++r)
      sH[wid][(r + kh3) * HH + n0 + col] = (h16)sspf(acc[r] + bias);
  }
  // GEMM2: Wf = (hidden @ w2 + b2) * C, then msg = x[row]*Wf -> atomic agg[col]
  v16h hf[4];
#pragma unroll
  for (int k = 0; k < 4; ++k) hf[k] = frag_h(sH[wid], HH, k * 32);
#pragma unroll
  for (int n0 = 0; n0 < HH; n0 += 16) {
    v8f acc = {};
#pragma unroll
    for (int k = 0; k < 4; ++k)
      acc = wmma32(hf[k], frag_h(sW2 + n0 * HH, HH, k * 32), acc);
    float bias = b2l[n0 + col];
#pragma unroll
    for (int r = 0; r < 8; ++r) {
      int m = r + kh3;
      float wfv = (acc[r] + bias) * sCv[wid][m];
      float msg = x[(size_t)sRow[wid][m] * HH + n0 + col] * wfv;
      atomicAdd(&agg[(size_t)sCol[wid][m] * HH + n0 + col], msg);
    }
  }
}

// t = ssp(agg @ lin2_w + lin2_b)  (f16 out)
__global__ void k_nu1(const float* __restrict__ agg, const h16* __restrict__ wf16,
                      const float* __restrict__ l2b, h16* __restrict__ tbuf, int layer) {
  __shared__ h16 sW[HH * HH];
  { const uint32_t* s = (const uint32_t*)(wf16 + (size_t)layer*WLAYER + W_L2T);
    uint32_t* d = (uint32_t*)sW;
    for (int i = threadIdx.x; i < HH*HH/2; i += blockDim.x) d[i] = s[i]; }
  __syncthreads();
  int wid = threadIdx.x >> 5;
  int t = blockIdx.x * 8 + wid;
  if (t >= NN / 16) return;
  const float* A = agg + (size_t)t * 16 * HH;
  v16h af[4];
#pragma unroll
  for (int k = 0; k < 4; ++k) af[k] = frag_f32(A, HH, k * 32);
  int col = lane_id() & 15;
  int kh3 = (lane_id() >> 4) << 3;
  const float* bl = l2b + layer * HH;
#pragma unroll
  for (int n0 = 0; n0 < HH; n0 += 16) {
    v8f acc = {};
#pragma unroll
    for (int k = 0; k < 4; ++k)
      acc = wmma32(af[k], frag_h(sW + n0 * HH, HH, k * 32), acc);
    float bias = bl[n0 + col];
#pragma unroll
    for (int r = 0; r < 8; ++r)
      tbuf[((size_t)t*16 + r + kh3) * HH + n0 + col] = (h16)sspf(acc[r] + bias);
  }
}

// h += t @ lin_w + lin_b
__global__ void k_nu2(const h16* __restrict__ tbuf, const h16* __restrict__ wf16,
                      const float* __restrict__ lb, float* __restrict__ h, int layer) {
  __shared__ h16 sW[HH * HH];
  { const uint32_t* s = (const uint32_t*)(wf16 + (size_t)layer*WLAYER + W_LT);
    uint32_t* d = (uint32_t*)sW;
    for (int i = threadIdx.x; i < HH*HH/2; i += blockDim.x) d[i] = s[i]; }
  __syncthreads();
  int wid = threadIdx.x >> 5;
  int t = blockIdx.x * 8 + wid;
  if (t >= NN / 16) return;
  const h16* A = tbuf + (size_t)t * 16 * HH;
  v16h af[4];
#pragma unroll
  for (int k = 0; k < 4; ++k) af[k] = frag_h(A, HH, k * 32);
  int col = lane_id() & 15;
  int kh3 = (lane_id() >> 4) << 3;
  const float* bl = lb + layer * HH;
#pragma unroll
  for (int n0 = 0; n0 < HH; n0 += 16) {
    v8f acc = {};
#pragma unroll
    for (int k = 0; k < 4; ++k)
      acc = wmma32(af[k], frag_h(sW + n0 * HH, HH, k * 32), acc);
    float bias = bl[n0 + col];
#pragma unroll
    for (int r = 0; r < 8; ++r) {
      size_t idx = ((size_t)t*16 + r + kh3) * HH + n0 + col;
      h[idx] += acc[r] + bias;
    }
  }
}

extern "C" void kernel_launch(void* const* d_in, const int* in_sizes, int n_in,
                              void* d_out, int out_size, void* d_ws, size_t ws_size,
                              hipStream_t stream) {
  const int*   z   = (const int*)  d_in[0];
  const float* pos = (const float*)d_in[1];
  const int*   ei  = (const int*)  d_in[2];
  const float* emb = (const float*)d_in[3];
  const float* w1  = (const float*)d_in[4];
  const float* b1  = (const float*)d_in[5];
  const float* w2  = (const float*)d_in[6];
  const float* b2  = (const float*)d_in[7];
  const float* l1w = (const float*)d_in[8];
  const float* l2w = (const float*)d_in[9];
  const float* l2b = (const float*)d_in[10];
  const float* lw  = (const float*)d_in[11];
  const float* lb  = (const float*)d_in[12];

  float* h  = (float*)d_out;                 // h lives in d_out, updated in place
  char*  ws = (char*)d_ws;
  h16*   gauss = (h16*)  (ws + O_GAUSS);
  float* Cb    = (float*)(ws + O_C);
  float* x     = (float*)(ws + O_X);
  float* agg   = (float*)(ws + O_AGG);
  h16*   tbuf  = (h16*)  (ws + O_T);
  h16*   wf16  = (h16*)  (ws + O_W);

  k_prep <<<EE/256,        256, 0, stream>>>(pos, ei, gauss, Cb);
  k_embed<<<(NN*HH)/256,   256, 0, stream>>>(z, emb, h);
  k_convw<<<(LL*WLAYER)/256, 256, 0, stream>>>(w1, w2, l1w, l2w, lw, wf16);

  const int nodeBlocks = (NN/16 + 7) / 8;    // 79
  for (int layer = 0; layer < LL; ++layer) {
    k_lin1<<<nodeBlocks, 256, 0, stream>>>(h, wf16, x, agg, layer);
    k_edge<<<EE/16/4,    128, 0, stream>>>(gauss, Cb, ei, x, wf16, b1, b2, agg, layer);
    k_nu1 <<<nodeBlocks, 256, 0, stream>>>(agg, wf16, l2b, tbuf, layer);
    k_nu2 <<<nodeBlocks, 256, 0, stream>>>(tbuf, wf16, lb, h, layer);
  }
}
